// GCNlayer_49108656063193
// MI455X (gfx1250) — matrix operand order repaired and checked by reading, not compiled, not run on hardware
//
#include <hip/hip_runtime.h>
#include <hip/hip_bf16.h>

typedef __attribute__((ext_vector_type(16))) _Float16 v16h;
typedef __attribute__((ext_vector_type(8)))  _Float16 v8h;
typedef __attribute__((ext_vector_type(8)))  float    v8f;
typedef __attribute__((ext_vector_type(4)))  int      v4i;

#define AS1 __attribute__((address_space(1)))
#define AS3 __attribute__((address_space(3)))

#define N_NODES  50000
#define N_EDGES  800000
#define N_GRAPHS 500
#define FOUT     256
#define BN_EPS   1e-5f
#define ROWS_PER_BLK 64
#define LDSPAD   4

#ifndef __has_builtin
#define __has_builtin(x) 0
#endif
#if __has_builtin(__builtin_amdgcn_global_load_async_to_lds_b128)
#define HAVE_ASYNC_LDS 1
#else
#define HAVE_ASYNC_LDS 0
#endif

__device__ __forceinline__ void atomAddF(float* p, float v) {
  unsafeAtomicAdd(p, v);   // -> global_atomic_add_f32 (no CAS loop)
}

// ---------------- degree / norm ----------------
__global__ void k_init_deg(float* deg) {
  int i = blockIdx.x * blockDim.x + threadIdx.x;
  if (i < N_NODES) deg[i] = 1.0f;                 // self loop contributes 1
}
__global__ void k_deg_edges(const int* __restrict__ dst, float* __restrict__ deg) {
  int e = blockIdx.x * blockDim.x + threadIdx.x;
  if (e < N_EDGES) atomAddF(&deg[dst[e]], 1.0f);
}
__global__ void k_rsqrt_inplace(float* d) {
  int i = blockIdx.x * blockDim.x + threadIdx.x;
  if (i < N_NODES) d[i] = rsqrtf(d[i]);           // dinv = deg^-1/2
}

// ---------------- per-layer weight prep: Wt[n][k] = (f16) W[k][n] ----------------
// Makes every B fragment load two contiguous 16-byte runs (2x global_load_b128 per tile).
__global__ void k_w_to_f16t(const float* __restrict__ W, _Float16* __restrict__ Wt, int fin) {
  const int idx = blockIdx.x * blockDim.x + threadIdx.x;
  if (idx < fin * FOUT) {
    const int k = idx / FOUT, n = idx % FOUT;
    Wt[(size_t)n * fin + k] = (_Float16)W[idx];
  }
}

// ---------------- WMMA GEMM: XW = H @ W  (H: N x FIN f32, Wt: 256 x FIN f16) --------
// block = 256 threads = 8 waves; block owns a 16-row M tile staged in LDS (async
// direct-to-LDS when available); each wave produces two 16x16 N tiles.
template<int FIN>
__global__ void k_gemm_wmma(const float* __restrict__ H, const _Float16* __restrict__ Wt,
                            float* __restrict__ XW) {
  constexpr int LDSP = FIN + LDSPAD;               // pad rows -> distinct banks per lane
  __shared__ float As[16 * LDSP];
  const int tid   = threadIdx.x;
  const int mbase = blockIdx.x * 16;

  // stage A tile (16 x FIN f32) into LDS
  for (int c = tid; c < 16 * (FIN / 4); c += 256) {
    const int r = c / (FIN / 4);
    const int o = (c % (FIN / 4)) * 4;
#if HAVE_ASYNC_LDS
    __builtin_amdgcn_global_load_async_to_lds_b128(
        (AS1 v4i*)(v4i*)(H + (size_t)(mbase + r) * FIN + o),
        (AS3 v4i*)(v4i*)(&As[r * LDSP + o]), 0, 0);
#else
    *(float4*)(&As[r * LDSP + o]) = *(const float4*)(H + (size_t)(mbase + r) * FIN + o);
#endif
  }
#if HAVE_ASYNC_LDS
#if __has_builtin(__builtin_amdgcn_s_wait_asynccnt)
  __builtin_amdgcn_s_wait_asynccnt(0);
#else
  asm volatile("s_wait_asynccnt 0x0" ::: "memory");
#endif
#endif
  __syncthreads();

  const int wave = tid >> 5, lane = tid & 31;
  const int m    = lane & 15;                     // A row within tile
  const int hi   = lane >> 4;                     // lane half select
  const int col  = lane & 15;                     // B/C column within tile
  const int n0   = wave * 32;
  const int n1   = n0 + 16;
  const _Float16* Bt0 = Wt + (size_t)(n0 + col) * FIN;
  const _Float16* Bt1 = Wt + (size_t)(n1 + col) * FIN;

  union HV { v16h v; v8h h[2]; };
  v8f c0 = {}, c1 = {};
  #pragma unroll
  for (int k0 = 0; k0 < FIN; k0 += 32) {
    HV a, b0, b1;
    #pragma unroll
    for (int e = 0; e < 16; ++e) {
      const int v = e >> 1, p = e & 1;
      // A 16x32 f16 layout: K = 16*(v>=4) + 8*(lane>=16) + 2*(v&3) + p
      const int ka = k0 + ((v >> 2) << 4) + (hi << 3) + ((v & 3) << 1) + p;
      a.v[e] = (_Float16)As[m * LDSP + ka];       // ds_load_b64 pairs + v_cvt_pk_f16_f32
    }
    // B 32x16 f16 layout: K = 16*(e>=8) + 8*(lane>=16) + (e&7)
    b0.h[0] = *(const v8h*)(Bt0 + k0 + hi * 8);
    b0.h[1] = *(const v8h*)(Bt0 + k0 + hi * 8 + 16);
    b1.h[0] = *(const v8h*)(Bt1 + k0 + hi * 8);
    b1.h[1] = *(const v8h*)(Bt1 + k0 + hi * 8 + 16);
    c0 = __builtin_amdgcn_wmma_f32_16x16x32_f16(false, a.v, false, b0.v, (short)0, c0, false, false);
    c1 = __builtin_amdgcn_wmma_f32_16x16x32_f16(false, a.v, false, b1.v, (short)0, c1, false, false);
  }

  // C/D layout: VGPR r -> M = r + 8*(lane>=16), N = lane&15
  const int rbase = mbase + (hi << 3);
  #pragma unroll
  for (int r = 0; r < 8; ++r) {
    XW[(size_t)(rbase + r) * FOUT + n0 + col] = c0[r];
    XW[(size_t)(rbase + r) * FOUT + n1 + col] = c1[r];
  }
}

// ---------------- self-loop init: AGG = dinv^2 * XW + b ----------------
__global__ void k_self_init(const float* __restrict__ XW, const float* __restrict__ dinv,
                            const float* __restrict__ bias, float* __restrict__ AGG) {
  const int row = blockIdx.x, t = threadIdx.x;
  const float di = dinv[row];
  AGG[(size_t)row * FOUT + t] = di * di * XW[(size_t)row * FOUT + t] + bias[t];
}

// ---------------- edge scatter: AGG[dst] += dinv[s]*dinv[d] * XW[src] ----------------
// one wave32 per edge iteration; float4 gathers; hw f32 atomics; prefetch next source row.
__global__ void k_scatter(const float* __restrict__ XW, const int* __restrict__ src,
                          const int* __restrict__ dst, const float* __restrict__ dinv,
                          float* __restrict__ AGG) {
  const int lane = threadIdx.x & 31;
  const int wave = blockIdx.x * (blockDim.x >> 5) + (threadIdx.x >> 5);
  const int nw   = gridDim.x * (blockDim.x >> 5);
  for (int e = wave; e < N_EDGES; e += nw) {
    const int s = src[e], d = dst[e];
    const int e2 = e + nw;
    if (e2 < N_EDGES)
      __builtin_prefetch(XW + (size_t)src[e2] * FOUT, 0, 1);   // global_prefetch_b8
    const float w = dinv[s] * dinv[d];
    const float4* sr = (const float4*)(XW + (size_t)s * FOUT);
    const float4 v0 = sr[lane];        // cols 4*lane   .. 4*lane+3
    const float4 v1 = sr[lane + 32];   // cols 128+4*lane ..
    float* dr = AGG + (size_t)d * FOUT;
    atomAddF(dr + 4 * lane + 0, w * v0.x);
    atomAddF(dr + 4 * lane + 1, w * v0.y);
    atomAddF(dr + 4 * lane + 2, w * v0.z);
    atomAddF(dr + 4 * lane + 3, w * v0.w);
    atomAddF(dr + 128 + 4 * lane + 0, w * v1.x);
    atomAddF(dr + 128 + 4 * lane + 1, w * v1.y);
    atomAddF(dr + 128 + 4 * lane + 2, w * v1.z);
    atomAddF(dr + 128 + 4 * lane + 3, w * v1.w);
  }
}

// ---------------- fused ReLU + per-feature sum/sumsq (thread = feature col) ----------
__global__ void k_relu_stats(float* __restrict__ H, float* __restrict__ stats) {
  const int t = threadIdx.x;
  const int r0 = blockIdx.x * ROWS_PER_BLK;
  const int r1 = (r0 + ROWS_PER_BLK < N_NODES) ? r0 + ROWS_PER_BLK : N_NODES;
  float s = 0.f, s2 = 0.f;
  for (int r = r0; r < r1; ++r) {
    float v = H[(size_t)r * FOUT + t];
    v = v > 0.f ? v : 0.f;
    H[(size_t)r * FOUT + t] = v;
    s += v; s2 += v * v;
  }
  atomAddF(&stats[t], s);
  atomAddF(&stats[FOUT + t], s2);
}

__global__ void k_relu_only(float* __restrict__ H) {
  const int t = threadIdx.x;
  const int r0 = blockIdx.x * ROWS_PER_BLK;
  const int r1 = (r0 + ROWS_PER_BLK < N_NODES) ? r0 + ROWS_PER_BLK : N_NODES;
  for (int r = r0; r < r1; ++r) {
    float v = H[(size_t)r * FOUT + t];
    H[(size_t)r * FOUT + t] = v > 0.f ? v : 0.f;
  }
}

// ---------------- BN apply: h = gamma*(h-mean)*rsqrt(var+eps)+beta ----------------
__global__ void k_bn_apply(float* __restrict__ H, const float* __restrict__ stats,
                           const float* __restrict__ gamma, const float* __restrict__ beta) {
  const int t = threadIdx.x;
  const float invN = 1.0f / (float)N_NODES;
  const float mean = stats[t] * invN;
  const float var  = stats[FOUT + t] * invN - mean * mean;   // biased, as torch
  const float sc   = gamma[t] * rsqrtf(var + BN_EPS);
  const float sh   = beta[t] - mean * sc;
  const int r0 = blockIdx.x * ROWS_PER_BLK;
  const int r1 = (r0 + ROWS_PER_BLK < N_NODES) ? r0 + ROWS_PER_BLK : N_NODES;
  for (int r = r0; r < r1; ++r)
    H[(size_t)r * FOUT + t] = H[(size_t)r * FOUT + t] * sc + sh;
}

// ---------------- global add pool ----------------
__global__ void k_pool(const float* __restrict__ H, const int* __restrict__ batch,
                       float* __restrict__ out) {
  const int t = threadIdx.x;
  const int r0 = blockIdx.x * ROWS_PER_BLK;
  const int r1 = (r0 + ROWS_PER_BLK < N_NODES) ? r0 + ROWS_PER_BLK : N_NODES;
  for (int r = r0; r < r1; ++r)
    atomAddF(&out[(size_t)batch[r] * FOUT + t], H[(size_t)r * FOUT + t]);
}

__global__ void k_zero(float* p, int n) {
  int i = blockIdx.x * blockDim.x + threadIdx.x;
  if (i < n) p[i] = 0.f;
}

// =====================================================================================
extern "C" void kernel_launch(void* const* d_in, const int* in_sizes, int n_in,
                              void* d_out, int out_size, void* d_ws, size_t ws_size,
                              hipStream_t stream) {
  (void)in_sizes; (void)n_in; (void)out_size; (void)ws_size;

  const float* x     = (const float*)d_in[0];
  const int*   ei    = (const int*)d_in[1];
  const int*   src   = ei;
  const int*   dst   = ei + N_EDGES;
  const int*   batch = (const int*)d_in[2];
  const float* W[6], *b[6], *gamma[5], *beta[5];
  for (int l = 0; l < 6; ++l) { W[l] = (const float*)d_in[3 + 2 * l];  b[l]    = (const float*)d_in[4 + 2 * l]; }
  for (int l = 0; l < 5; ++l) { gamma[l] = (const float*)d_in[15 + 2 * l]; beta[l] = (const float*)d_in[16 + 2 * l]; }

  const size_t NF = (size_t)N_NODES * FOUT;   // 12.8M floats
  float* wsf   = (float*)d_ws;
  float* dinv  = wsf;                          // 50000
  float* stats = wsf + 50176;                  // 512 (16B aligned)
  float* xw    = wsf + 50176 + 512;            // N x 256
  float* bufA  = xw + NF;
  float* bufB  = bufA + NF;
  _Float16* wt[6];
  {
    _Float16* p = (_Float16*)(bufB + NF);
    int fins[6] = {128, 256, 256, 256, 256, 256};
    for (int l = 0; l < 6; ++l) { wt[l] = p; p += (size_t)FOUT * fins[l]; }
  }

  const int rowsGrid = (N_NODES + ROWS_PER_BLK - 1) / ROWS_PER_BLK;   // 782

  // degree -> dinv ; weight conversion (transposed f16)
  k_init_deg <<<(N_NODES + 255) / 256, 256, 0, stream>>>(dinv);
  k_deg_edges<<<(N_EDGES + 255) / 256, 256, 0, stream>>>(dst, dinv);
  k_rsqrt_inplace<<<(N_NODES + 255) / 256, 256, 0, stream>>>(dinv);
  k_w_to_f16t<<<(128 * FOUT + 255) / 256, 256, 0, stream>>>(W[0], wt[0], 128);
  for (int l = 1; l < 6; ++l)
    k_w_to_f16t<<<(256 * FOUT + 255) / 256, 256, 0, stream>>>(W[l], wt[l], 256);

  const float* h = x;
  float* cur = bufA;
  float* nxt = bufB;
  for (int l = 0; l < 6; ++l) {
    if (l == 0) k_gemm_wmma<128><<<N_NODES / 16, 256, 0, stream>>>(h, wt[l], xw);
    else        k_gemm_wmma<256><<<N_NODES / 16, 256, 0, stream>>>(h, wt[l], xw);
    k_self_init<<<N_NODES, 256, 0, stream>>>(xw, dinv, b[l], cur);
    k_scatter  <<<4096,    256, 0, stream>>>(xw, src, dst, dinv, cur);
    if (l < 5) {
      k_zero      <<<2, 256, 0, stream>>>(stats, 2 * FOUT);
      k_relu_stats<<<rowsGrid, 256, 0, stream>>>(cur, stats);
      k_bn_apply  <<<rowsGrid, 256, 0, stream>>>(cur, stats, gamma[l], beta[l]);
    } else {
      k_relu_only <<<rowsGrid, 256, 0, stream>>>(cur);
    }
    h = cur;
    float* tmp = cur; cur = nxt; nxt = tmp;
  }

  k_zero<<<(N_GRAPHS * FOUT + 255) / 256, 256, 0, stream>>>((float*)d_out, N_GRAPHS * FOUT);
  k_pool<<<rowsGrid, 256, 0, stream>>>(h, batch, (float*)d_out);
}
